// SSMDecoder_14602888806691
// MI455X (gfx1250) — matrix operand order repaired and checked by reading, not compiled, not run on hardware
//
#include <hip/hip_runtime.h>
#include <hip/hip_bf16.h>
#include <stdint.h>

// ---------------------------------------------------------------------------
// SSM decoder (Mamba-style) for MI455X / gfx1250.
// All dense GEMMs run through v_wmma_f32_16x16x32_bf16 (fp32 accumulate).
// Large GEMMs: 32x32 wave tile (4 WMMAs / 8 b128 loads per K-step), sized to
// avoid VGPR spills under the compiler's software pipelining.
// Sequential selective scan: one wave32 per channel, lane = state.
// ---------------------------------------------------------------------------

#define LSEQ 2048
#define DDIM 1024
#define NSTATE 32
#define VOCAB 32000

typedef __attribute__((ext_vector_type(16))) __bf16 bf16x16;
typedef __attribute__((ext_vector_type(8)))  float  floatx8;

union FragAB {
    uint4   q[2];   // 32 bytes = 16 bf16
    bf16x16 v;
};

// ---------------------------- helpers --------------------------------------

__device__ __forceinline__ unsigned short f32_to_bf16_rne(float f) {
    unsigned int u = __float_as_uint(f);
    unsigned int r = (u + 0x7FFFu + ((u >> 16) & 1u)) >> 16;   // round-to-nearest-even
    return (unsigned short)r;
}

// ------------------------- convert f32 -> bf16 ------------------------------

__global__ void cvt_f32_bf16_kernel(const float* __restrict__ src,
                                    unsigned short* __restrict__ dst,
                                    size_t n) {
    size_t i = (size_t)blockIdx.x * blockDim.x + threadIdx.x;
    size_t stride = (size_t)gridDim.x * blockDim.x;
    for (; i < n; i += stride) dst[i] = f32_to_bf16_rne(src[i]);
}

// ------------------- blocked WMMA GEMM (large N) ----------------------------
// C[M x N] = epilogue( A[M x K](bf16) * B[N x K]^T(bf16) + bias[N] )
// Requires M % 64 == 0, N % 128 == 0, K % 32 == 0.
// Block: 256 threads = 8 waves tiling 64(M) x 128(N) of C.
// Wave tile: 32 x 32 -> 2 A-frags x 2 B-frags -> 4 accumulators (32 VGPRs);
// fits the 256-VGPR window with the compiler's pipelined loads (no spills).

__global__ void wmma_gemm_blk_kernel(const unsigned short* __restrict__ A,
                                     const unsigned short* __restrict__ B,
                                     const float* __restrict__ bias,
                                     float* __restrict__ C,
                                     int M, int Nn, int K, int mode) {
    const int wave = threadIdx.x >> 5;
    const int lane = threadIdx.x & 31;
    const int m_base = blockIdx.y * 64 + (wave >> 2) * 32;    // wave rows
    const int n_base = blockIdx.x * 128 + (wave & 3) * 32;    // wave cols
    const int half = lane >> 4;
    const int l16  = lane & 15;

    const uint4* Arow0 = reinterpret_cast<const uint4*>(A + (size_t)(m_base + l16) * K);
    const uint4* Arow1 = reinterpret_cast<const uint4*>(A + (size_t)(m_base + 16 + l16) * K);
    const uint4* Brow0 = reinterpret_cast<const uint4*>(B + (size_t)(n_base + l16) * K);
    const uint4* Brow1 = reinterpret_cast<const uint4*>(B + (size_t)(n_base + 16 + l16) * K);

    floatx8 acc00 = {}, acc01 = {}, acc10 = {}, acc11 = {};

    for (int k0 = 0; k0 < K; k0 += 32) {
        // A 16x32 bf16 frag: elems 0-7 -> K=k0+8*half+i; 8-15 -> K=k0+16+8*half+(i-8)
        const int aq0 = (k0 + 8 * half) >> 3;
        const int aq1 = (k0 + 16 + 8 * half) >> 3;
        // B 32x16 bf16 frag: same K order as A
        const int bq = (k0 + 8 * half) >> 3;
        const int bq1 = (k0 + 16 + 8 * half) >> 3;

        FragAB a0, a1, b0, b1;
        a0.q[0] = Arow0[aq0]; a0.q[1] = Arow0[aq1];
        a1.q[0] = Arow1[aq0]; a1.q[1] = Arow1[aq1];
        b0.q[0] = Brow0[bq];  b0.q[1] = Brow0[bq1];
        b1.q[0] = Brow1[bq];  b1.q[1] = Brow1[bq1];

        acc00 = __builtin_amdgcn_wmma_f32_16x16x32_bf16(false, a0.v, false, b0.v, (short)0, acc00, false, false);
        acc01 = __builtin_amdgcn_wmma_f32_16x16x32_bf16(false, a0.v, false, b1.v, (short)0, acc01, false, false);
        acc10 = __builtin_amdgcn_wmma_f32_16x16x32_bf16(false, a1.v, false, b0.v, (short)0, acc10, false, false);
        acc11 = __builtin_amdgcn_wmma_f32_16x16x32_bf16(false, a1.v, false, b1.v, (short)0, acc11, false, false);
    }

    // Epilogue: C element r of lane -> row (m + r + 8*half), col (n + l16).
    const int col0 = n_base + l16;
    const int col1 = n_base + 16 + l16;
    const float bv0 = bias ? bias[col0] : 0.0f;
    const float bv1 = bias ? bias[col1] : 0.0f;
#pragma unroll
    for (int r = 0; r < 8; ++r) {
        const size_t row0 = (size_t)(m_base + r + 8 * half) * Nn;
        const size_t row1 = (size_t)(m_base + 16 + r + 8 * half) * Nn;
        float v00 = acc00[r] + bv0, v01 = acc01[r] + bv1;
        float v10 = acc10[r] + bv0, v11 = acc11[r] + bv1;
        if (mode == 1) {
            v00 = (v00 > 20.0f) ? v00 : log1pf(__expf(v00));   // softplus
            v01 = (v01 > 20.0f) ? v01 : log1pf(__expf(v01));
            v10 = (v10 > 20.0f) ? v10 : log1pf(__expf(v10));
            v11 = (v11 > 20.0f) ? v11 : log1pf(__expf(v11));
        }
        C[row0 + col0] = v00; C[row0 + col1] = v01;
        C[row1 + col0] = v10; C[row1 + col1] = v11;
    }
}

// ------------------- simple WMMA GEMM (small N, e.g. 32) --------------------
// One wave per 16x16 tile, 4 waves per block along N.

__global__ void wmma_gemm_kernel(const unsigned short* __restrict__ A,
                                 const unsigned short* __restrict__ B,
                                 const float* __restrict__ bias,
                                 float* __restrict__ C,
                                 int M, int Nn, int K, int mode) {
    const int wave = threadIdx.x >> 5;
    const int lane = threadIdx.x & 31;
    const int n0 = (blockIdx.x * 4 + wave) * 16;
    const int m0 = blockIdx.y * 16;
    if (n0 >= Nn || m0 >= M) return;

    const int half = lane >> 4;
    const int l16  = lane & 15;

    const uint4* Arow = reinterpret_cast<const uint4*>(A + (size_t)(m0 + l16) * K);
    const uint4* Brow = reinterpret_cast<const uint4*>(B + (size_t)(n0 + l16) * K);

    floatx8 acc = {};
    for (int k0 = 0; k0 < K; k0 += 32) {
        FragAB a, b;
        a.q[0] = Arow[(k0 + 8 * half) >> 3];
        a.q[1] = Arow[(k0 + 16 + 8 * half) >> 3];
        b.q[0] = Brow[(k0 + 8 * half) >> 3];
        b.q[1] = Brow[(k0 + 16 + 8 * half) >> 3];
        acc = __builtin_amdgcn_wmma_f32_16x16x32_bf16(
            false, a.v, false, b.v, (short)0, acc, false, false);
    }

    const float bv = bias ? bias[n0 + l16] : 0.0f;
    const int col = n0 + l16;
#pragma unroll
    for (int r = 0; r < 8; ++r) {
        float v = acc[r] + bv;
        if (mode == 1) v = (v > 20.0f) ? v : log1pf(__expf(v));
        C[(size_t)(m0 + r + 8 * half) * Nn + col] = v;
    }
}

// --------------- split xz -> x_in (f32 + bf16), z = sigmoid ------------------

__global__ void split_xz_kernel(const float* __restrict__ xz,
                                float* __restrict__ x_in_f,
                                unsigned short* __restrict__ x_in_bf,
                                float* __restrict__ z) {
    size_t i = (size_t)blockIdx.x * blockDim.x + threadIdx.x;
    size_t stride = (size_t)gridDim.x * blockDim.x;
    const size_t total = (size_t)LSEQ * DDIM;
    for (; i < total; i += stride) {
        size_t r = i / DDIM, c = i - r * DDIM;
        float a = xz[r * (2 * DDIM) + c];
        float b = xz[r * (2 * DDIM) + DDIM + c];
        x_in_f[i] = a;
        x_in_bf[i] = f32_to_bf16_rne(a);
        z[i] = 1.0f / (1.0f + __expf(-b));
    }
}

// ------------------- combined delta bias (b_delta + delta_bias) -------------

__global__ void add_bias_kernel(const float* __restrict__ a,
                                const float* __restrict__ b,
                                float* __restrict__ o, int n) {
    int i = blockIdx.x * blockDim.x + threadIdx.x;
    if (i < n) o[i] = a[i] + b[i];
}

// ------------- injection gates: Bg = xb*(1+tanh(gb)), Cg likewise ------------

__global__ void gate_kernel(const float* __restrict__ xb,
                            const float* __restrict__ gb,
                            const float* __restrict__ xc,
                            const float* __restrict__ gc,
                            float* __restrict__ Bg,
                            float* __restrict__ Cg) {
    size_t i = (size_t)blockIdx.x * blockDim.x + threadIdx.x;
    size_t stride = (size_t)gridDim.x * blockDim.x;
    const size_t total = (size_t)LSEQ * NSTATE;
    for (; i < total; i += stride) {
        Bg[i] = xb[i] * (1.0f + tanhf(gb[i]));
        Cg[i] = xc[i] * (1.0f + tanhf(gc[i]));
    }
}

// --------------------------- selective scan ---------------------------------

__global__ void scan_kernel(const float* __restrict__ logA,   // D x N
                            const float* __restrict__ delta,  // L x D
                            const float* __restrict__ Bg,     // L x N
                            const float* __restrict__ Cg,     // L x N
                            float* __restrict__ y) {          // L x D
    const int d = blockIdx.x * (blockDim.x >> 5) + (threadIdx.x >> 5);
    const int n = threadIdx.x & 31;
    if (d >= DDIM) return;

    const float Aval = -__expf(logA[d * NSTATE + n]);
    const float invA = 1.0f / Aval;
    float h = 0.0f;

    for (int t = 0; t < LSEQ; ++t) {
        const float dt = delta[(size_t)t * DDIM + d];
        const float ab = __expf(dt * Aval);
        const float bb = (ab - 1.0f) * invA * Bg[t * NSTATE + n];
        h = fmaf(ab, h, bb);
        float p = h * Cg[t * NSTATE + n];
#pragma unroll
        for (int o = 16; o > 0; o >>= 1) p += __shfl_xor(p, o, 32);
        if (n == 0) y[(size_t)t * DDIM + d] = p;
    }
}

// --------------- u = y*z + x_in ; LayerNorm(u) -> bf16 ----------------------

__global__ void ln_kernel(const float* __restrict__ y,
                          const float* __restrict__ z,
                          const float* __restrict__ xin,
                          const float* __restrict__ g,
                          const float* __restrict__ b,
                          unsigned short* __restrict__ out_bf) {
    __shared__ float s1[256], s2[256];
    const int row = blockIdx.x;
    const int tid = threadIdx.x;
    const size_t base = (size_t)row * DDIM;

    float u[4], s = 0.0f, ss = 0.0f;
#pragma unroll
    for (int j = 0; j < 4; ++j) {
        int c = tid + j * 256;
        float v = fmaf(y[base + c], z[base + c], xin[base + c]);
        u[j] = v; s += v; ss += v * v;
    }
    s1[tid] = s; s2[tid] = ss;
    __syncthreads();
    for (int o = 128; o > 0; o >>= 1) {
        if (tid < o) { s1[tid] += s1[tid + o]; s2[tid] += s2[tid + o]; }
        __syncthreads();
    }
    const float mean = s1[0] * (1.0f / DDIM);
    const float var  = s2[0] * (1.0f / DDIM) - mean * mean;
    const float rstd = rsqrtf(var + 1e-5f);
#pragma unroll
    for (int j = 0; j < 4; ++j) {
        int c = tid + j * 256;
        float o = (u[j] - mean) * rstd * g[c] + b[c];
        out_bf[base + c] = f32_to_bf16_rne(o);
    }
}

// ---------------------------------------------------------------------------

extern "C" void kernel_launch(void* const* d_in, const int* in_sizes, int n_in,
                              void* d_out, int out_size, void* d_ws, size_t ws_size,
                              hipStream_t stream) {
    (void)in_sizes; (void)n_in; (void)out_size; (void)ws_size;

    const float* x        = (const float*)d_in[0];
    const float* b_inject = (const float*)d_in[1];
    const float* c_inject = (const float*)d_in[2];
    const float* W_in     = (const float*)d_in[3];
    const float* b_in     = (const float*)d_in[4];
    const float* log_A    = (const float*)d_in[5];
    const float* W_B      = (const float*)d_in[6];
    const float* W_C      = (const float*)d_in[7];
    const float* W_delta  = (const float*)d_in[8];
    const float* b_delta  = (const float*)d_in[9];
    const float* dlt_bias = (const float*)d_in[10];
    const float* W_bi     = (const float*)d_in[11];
    const float* W_ci     = (const float*)d_in[12];
    const float* ln_g     = (const float*)d_in[13];
    const float* ln_b     = (const float*)d_in[14];
    const float* W_out    = (const float*)d_in[15];
    const float* b_out    = (const float*)d_in[16];
    const float* W_vocab  = (const float*)d_in[17];
    const float* b_vocab  = (const float*)d_in[18];
    float* logits = (float*)d_out;

    // ------------------ workspace bump allocator (256B aligned) -------------
    char* wp = (char*)d_ws;
    auto alloc = [&](size_t bytes) -> char* {
        char* p = wp; wp += (bytes + 255) & ~(size_t)255; return p;
    };
    const size_t LD4 = (size_t)LSEQ * DDIM * 4, LD2 = (size_t)LSEQ * DDIM * 2;

    unsigned short* x_bf     = (unsigned short*)alloc(LD2);
    unsigned short* Win_bf   = (unsigned short*)alloc((size_t)2 * DDIM * DDIM * 2);
    float*          xz       = (float*)alloc((size_t)LSEQ * 2 * DDIM * 4);
    float*          x_in_f   = (float*)alloc(LD4);
    unsigned short* x_in_bf  = (unsigned short*)alloc(LD2);
    float*          z        = (float*)alloc(LD4);
    unsigned short* Wd_bf    = (unsigned short*)alloc((size_t)DDIM * DDIM * 2);
    float*          dbc      = (float*)alloc(DDIM * 4);
    float*          delta    = (float*)alloc(LD4);
    unsigned short* bj_bf    = (unsigned short*)alloc(LD2);
    unsigned short* cj_bf    = (unsigned short*)alloc(LD2);
    unsigned short* WB_bf    = (unsigned short*)alloc((size_t)NSTATE * DDIM * 2);
    unsigned short* WC_bf    = (unsigned short*)alloc((size_t)NSTATE * DDIM * 2);
    unsigned short* Wbi_bf   = (unsigned short*)alloc((size_t)NSTATE * DDIM * 2);
    unsigned short* Wci_bf   = (unsigned short*)alloc((size_t)NSTATE * DDIM * 2);
    float*          xb       = (float*)alloc((size_t)LSEQ * NSTATE * 4);
    float*          gb       = (float*)alloc((size_t)LSEQ * NSTATE * 4);
    float*          xc       = (float*)alloc((size_t)LSEQ * NSTATE * 4);
    float*          gc       = (float*)alloc((size_t)LSEQ * NSTATE * 4);
    float*          Bg       = (float*)alloc((size_t)LSEQ * NSTATE * 4);
    float*          Cg       = (float*)alloc((size_t)LSEQ * NSTATE * 4);
    unsigned short* Wv_bf    = (unsigned short*)alloc((size_t)VOCAB * DDIM * 2);
    // phase reuse (producers dead before consumers run):
    float*          yb       = xz;                               // L x D
    unsigned short* ln_bf    = x_bf;                             // L x D bf16
    unsigned short* Wout_bf  = Win_bf;                           // D x D bf16
    float*          out_f    = (float*)((char*)xz + LD4);        // L x D
    unsigned short* out_bf   = bj_bf;                            // L x D bf16

    const int CT = 256, CG = 2048;
    auto cvt = [&](const float* s, unsigned short* d, size_t n) {
        cvt_f32_bf16_kernel<<<CG, CT, 0, stream>>>(s, d, n);
    };
    auto gemm = [&](const unsigned short* A, const unsigned short* B,
                    const float* bias, float* C, int M, int Nn, int K, int mode) {
        if ((Nn % 128 == 0) && (M % 64 == 0)) {
            dim3 grid(Nn / 128, M / 64);
            wmma_gemm_blk_kernel<<<grid, 256, 0, stream>>>(A, B, bias, C, M, Nn, K, mode);
        } else {
            dim3 grid((Nn + 63) / 64, M / 16);
            wmma_gemm_kernel<<<grid, 128, 0, stream>>>(A, B, bias, C, M, Nn, K, mode);
        }
    };

    // 1) xz = x @ W_in^T + b_in
    cvt(x, x_bf, (size_t)LSEQ * DDIM);
    cvt(W_in, Win_bf, (size_t)2 * DDIM * DDIM);
    gemm(x_bf, Win_bf, b_in, xz, LSEQ, 2 * DDIM, DDIM, 0);

    // 2) split into x_in (f32 + bf16) and z = sigmoid(...)
    split_xz_kernel<<<CG, CT, 0, stream>>>(xz, x_in_f, x_in_bf, z);

    // 3) delta = softplus(x_in @ W_delta^T + b_delta + delta_bias)
    cvt(W_delta, Wd_bf, (size_t)DDIM * DDIM);
    add_bias_kernel<<<DDIM / 256, 256, 0, stream>>>(b_delta, dlt_bias, dbc, DDIM);
    gemm(x_in_bf, Wd_bf, dbc, delta, LSEQ, DDIM, DDIM, 1);

    // 4) gated B/C sequences (N = 32 -> simple kernel)
    cvt(b_inject, bj_bf, (size_t)LSEQ * DDIM);
    cvt(c_inject, cj_bf, (size_t)LSEQ * DDIM);
    cvt(W_B,  WB_bf,  (size_t)NSTATE * DDIM);
    cvt(W_C,  WC_bf,  (size_t)NSTATE * DDIM);
    cvt(W_bi, Wbi_bf, (size_t)NSTATE * DDIM);
    cvt(W_ci, Wci_bf, (size_t)NSTATE * DDIM);
    gemm(x_in_bf, WB_bf,  nullptr, xb, LSEQ, NSTATE, DDIM, 0);
    gemm(bj_bf,   Wbi_bf, nullptr, gb, LSEQ, NSTATE, DDIM, 0);
    gemm(x_in_bf, WC_bf,  nullptr, xc, LSEQ, NSTATE, DDIM, 0);
    gemm(cj_bf,   Wci_bf, nullptr, gc, LSEQ, NSTATE, DDIM, 0);
    gate_kernel<<<256, 256, 0, stream>>>(xb, gb, xc, gc, Bg, Cg);

    // 5) selective scan -> y
    scan_kernel<<<DDIM / 4, 128, 0, stream>>>(log_A, delta, Bg, Cg, yb);

    // 6) u = y*z + x_in ; LayerNorm -> bf16
    ln_kernel<<<LSEQ, 256, 0, stream>>>(yb, z, x_in_f, ln_g, ln_b, ln_bf);

    // 7) out = ln @ W_out^T + b_out
    cvt(W_out, Wout_bf, (size_t)DDIM * DDIM);
    gemm(ln_bf, Wout_bf, b_out, out_f, LSEQ, DDIM, DDIM, 0);
    cvt(out_f, out_bf, (size_t)LSEQ * DDIM);

    // 8) logits = out @ W_vocab^T + b_vocab   (dominant GEMM, 134 GFLOP)
    cvt(W_vocab, Wv_bf, (size_t)VOCAB * DDIM);
    gemm(out_bf, Wv_bf, b_vocab, logits, LSEQ, VOCAB, DDIM, 0);
}
